// VisionTransformer_4054449127636
// MI455X (gfx1250) — matrix operand order repaired and checked by reading, not compiled, not run on hardware
//
#include <hip/hip_runtime.h>
#include <hip/hip_bf16.h>
#include <math.h>

typedef __bf16 bf16;
typedef __attribute__((ext_vector_type(8)))  bf16  v8bf;
typedef __attribute__((ext_vector_type(16))) bf16  v16bf;
typedef __attribute__((ext_vector_type(8)))  float v8f;
typedef __attribute__((ext_vector_type(4)))  unsigned int u32x4;
typedef __attribute__((ext_vector_type(8)))  int   i32x8;
typedef __attribute__((ext_vector_type(4)))  int   i32x4;

#define B_    64
#define D_    768
#define N_    197
#define NP_   196
#define HEADS_ 12
#define HD_   64
#define HID_  3072
#define DEPTH_ 12
#define M_    (B_ * N_)    // 12608 = 64*197
#define MP_   (B_ * NP_)   // 12544 = 64*196

#if __has_builtin(__builtin_amdgcn_tensor_load_to_lds) && \
    __has_builtin(__builtin_amdgcn_s_wait_tensorcnt)
#define HAS_TDM 1
#else
#define HAS_TDM 0
#endif

static __device__ __forceinline__ v16bf cat8(v8bf lo, v8bf hi) {
  return __builtin_shufflevector(lo, hi, 0,1,2,3,4,5,6,7,8,9,10,11,12,13,14,15);
}
static __device__ __forceinline__ v8f wmma_bf16(v16bf a, v16bf b, v8f c) {
  return __builtin_amdgcn_wmma_f32_16x16x32_bf16(false, a, false, b, (short)0, c, false, false);
}

#if HAS_TDM
// Issue one TDM 2D tile load: bf16 tensor, row-major, tile_d0 (K) x tile_d1 (rows),
// row stride = tensor_d0 elements. LDS gets +16B pad per 64B row (bank-spread),
// matching the [rows][40] compute-side layout when tile_d0 == 32.
static __device__ __forceinline__ void tdm_load_tile(const bf16* gptr, unsigned lds_off,
                                                     unsigned tensor_d0, unsigned tile_d0,
                                                     unsigned tile_d1) {
  unsigned long long ga = (unsigned long long)(uintptr_t)gptr;
  u32x4 g0;
  g0[0] = 1u;                                   // count=1, user descriptor
  g0[1] = lds_off;                              // lds_addr (bytes)
  g0[2] = (unsigned)ga;                         // global_addr[31:0]
  g0[3] = (unsigned)(ga >> 32) | (2u << 30);    // global_addr[56:32] | type=2 (image)
  const unsigned tensor_d1 = 1u << 20;          // generous row extent (tiles in-bounds)
  i32x8 g1;
  g1[0] = (int)((1u << 16)          // data_size = 2 bytes
              | (1u << 20)          // pad_enable
              | (3u << 22)          // pad_interval: 64 bytes stored ...
              | (3u << 25));        // pad_amount: ... then 16 bytes pad
  g1[1] = (int)((tensor_d0 & 0xFFFFu) << 16);                    // abar=0 | dim0 lo16
  g1[2] = (int)((tensor_d0 >> 16) | ((tensor_d1 & 0xFFFFu) << 16)); // dim0 hi | dim1 lo
  g1[3] = (int)((tensor_d1 >> 16) | (tile_d0 << 16));            // dim1 hi | tile_dim0
  g1[4] = (int)tile_d1;                                          // tile_dim1 | tile_dim2=0
  g1[5] = (int)tensor_d0;                                        // dim0_stride lo32
  g1[6] = 0;                                                     // stride hi | dim1_stride lo
  g1[7] = 0;
  i32x4 z4 = {0, 0, 0, 0};
#if defined(__clang_major__) && __clang_major__ >= 23
  i32x8 z8 = {0, 0, 0, 0, 0, 0, 0, 0};
  __builtin_amdgcn_tensor_load_to_lds(g0, g1, z4, z4, z8, 0);
#else
  __builtin_amdgcn_tensor_load_to_lds(g0, g1, z4, z4, 0);
#endif
}
#endif

// ---------------- elementwise helpers ----------------

__global__ void cvt_bf16(const float* __restrict__ s, bf16* __restrict__ d, int n) {
  for (int i = blockIdx.x * blockDim.x + threadIdx.x; i < n; i += gridDim.x * blockDim.x)
    d[i] = (bf16)s[i];
}

// im2col for 16x16 stride-16 patches: A[b*196 + py*14 + px][c*256 + i*16 + j]
__global__ void im2col_patch(const float* __restrict__ x, bf16* __restrict__ a) {
  const int total = MP_ * D_;
  for (int idx = blockIdx.x * blockDim.x + threadIdx.x; idx < total;
       idx += gridDim.x * blockDim.x) {
    int row = idx / D_, k = idx - row * D_;
    int b = row / NP_, p = row - b * NP_;
    int py = p / 14, px = p - py * 14;
    int c = k >> 8, i = (k >> 4) & 15, j = k & 15;
    a[idx] = (bf16)x[(((b * 3 + c) * 224) + py * 16 + i) * 224 + px * 16 + j];
  }
}

// h[b, n, d] = (n==0 ? cls[d] : tmp[(b*196 + n-1)*768 + d]) + pos[n*768 + d]
__global__ void assemble_tokens(const float* __restrict__ tmp, const float* __restrict__ cls,
                                const float* __restrict__ pos, float* __restrict__ h) {
  const int total = M_ * D_;
  for (int idx = blockIdx.x * blockDim.x + threadIdx.x; idx < total;
       idx += gridDim.x * blockDim.x) {
    int d = idx % D_;
    int bn = idx / D_;
    int n = bn % N_;
    float v = (n == 0) ? cls[d] : tmp[((bn / N_) * NP_ + n - 1) * D_ + d];
    h[idx] = v + pos[n * D_ + d];
  }
}

// ---------------- layernorm: one wave per row of 768 ----------------

template <int OUTB>
__global__ __launch_bounds__(256) void layernorm_k(const float* __restrict__ in,
                                                   const float* __restrict__ wt,
                                                   const float* __restrict__ bs,
                                                   bf16* __restrict__ outb,
                                                   float* __restrict__ outf) {
  const int w = threadIdx.x >> 5, lane = threadIdx.x & 31;
  const long row = (long)blockIdx.x * 8 + w;
  const float* p = in + row * D_;
  float x[24];
  float s = 0.f;
#pragma unroll
  for (int i = 0; i < 24; ++i) { x[i] = p[i * 32 + lane]; s += x[i]; }
#pragma unroll
  for (int off = 16; off; off >>= 1) s += __shfl_xor(s, off, 32);
  const float mu = s * (1.0f / 768.0f);
  float v = 0.f;
#pragma unroll
  for (int i = 0; i < 24; ++i) { float d = x[i] - mu; v += d * d; }
#pragma unroll
  for (int off = 16; off; off >>= 1) v += __shfl_xor(v, off, 32);
  const float rs = rsqrtf(v * (1.0f / 768.0f) + 1e-6f);
#pragma unroll
  for (int i = 0; i < 24; ++i) {
    int c = i * 32 + lane;
    float y = (x[i] - mu) * rs * wt[c] + bs[c];
    if (OUTB) outb[row * D_ + c] = (bf16)y;
    else      outf[row * D_ + c] = y;
  }
}

// ---------------- bf16 WMMA GEMM: C = A[M,K] @ W[N,K]^T + bias, epilogues ----------------
// EPI: 0 = bf16 out, 1 = bf16 out + exact GELU, 2 = f32 out, 3 = f32 out + residual
// Block 256 threads = 8 waves; tile 64(M) x 128(N), BK=32; wave w owns n-tile w (16 cols).
// TDM path: wave 0 issues double-buffered tensor_load_to_lds for next K-step while
// all waves run WMMA on the current buffer; synced via s_wait_tensorcnt + barrier.

template <int EPI>
__global__ __launch_bounds__(256) void gemm_bf16_k(const bf16* __restrict__ A,
                                                   const bf16* __restrict__ W,
                                                   const float* __restrict__ bias,
                                                   const float* __restrict__ res,
                                                   float* __restrict__ outf,
                                                   bf16* __restrict__ outb,
                                                   int K, int Nout) {
  constexpr int BK = 32, PAD = 8, LDW = BK + PAD;
  __shared__ bf16 As[2][64][LDW];
  __shared__ bf16 Ws[2][128][LDW];
  const int tid = threadIdx.x;
  const int w = tid >> 5, lane = tid & 31;
  const int lh = lane >> 4, l16 = lane & 15;
  const long m0 = (long)blockIdx.x * 64;
  const long n0 = (long)blockIdx.y * 128;

  v8f acc[4] = {};

#if HAS_TDM
  const int nk = K / BK;
  if (w == 0) {
    tdm_load_tile(A + m0 * K, (unsigned)(uintptr_t)&As[0][0][0], K, BK, 64);
    tdm_load_tile(W + n0 * K, (unsigned)(uintptr_t)&Ws[0][0][0], K, BK, 128);
  }
  for (int i = 0; i < nk; ++i) {
    const int cur = i & 1;
    if (w == 0) {
      if (i + 1 < nk) {
        tdm_load_tile(A + m0 * K + (i + 1) * BK, (unsigned)(uintptr_t)&As[1 - cur][0][0],
                      K, BK, 64);
        tdm_load_tile(W + n0 * K + (i + 1) * BK, (unsigned)(uintptr_t)&Ws[1 - cur][0][0],
                      K, BK, 128);
        __builtin_amdgcn_s_wait_tensorcnt(2);  // current buffer's pair has landed
      } else {
        __builtin_amdgcn_s_wait_tensorcnt(0);
      }
    }
    __syncthreads();
    const bf16* wsrow = &Ws[cur][w * 16 + l16][lh * 16];
    v16bf bfrag = cat8(*(const v8bf*)wsrow, *(const v8bf*)(wsrow + 8));
#pragma unroll
    for (int mt = 0; mt < 4; ++mt) {
      const bf16* asrow = &As[cur][mt * 16 + l16][lh * 16];
      v16bf afrag = cat8(*(const v8bf*)asrow, *(const v8bf*)(asrow + 8));
      acc[mt] = wmma_bf16(afrag, bfrag, acc[mt]);
    }
    __syncthreads();
  }
#else
  const int ar = tid >> 2, ac = (tid & 3) * 8;   // A stage: 64 rows x 32, 8 elems/thread
  const int wr = tid >> 1, wc = (tid & 1) * 16;  // W stage: 128 rows x 32, 16 elems/thread
  for (int k0 = 0; k0 < K; k0 += BK) {
    *(v8bf*)&As[0][ar][ac]     = *(const v8bf*)&A[(m0 + ar) * K + k0 + ac];
    *(v8bf*)&Ws[0][wr][wc]     = *(const v8bf*)&W[(n0 + wr) * K + k0 + wc];
    *(v8bf*)&Ws[0][wr][wc + 8] = *(const v8bf*)&W[(n0 + wr) * K + k0 + wc + 8];
    if (k0 + BK < K) {
      __builtin_prefetch(&A[(m0 + ar) * K + k0 + BK + ac], 0, 3);
      __builtin_prefetch(&W[(n0 + wr) * K + k0 + BK + wc], 0, 3);
    }
    __syncthreads();
    const bf16* wsrow = &Ws[0][w * 16 + l16][lh * 16];
    v16bf bfrag = cat8(*(const v8bf*)wsrow, *(const v8bf*)(wsrow + 8));
#pragma unroll
    for (int mt = 0; mt < 4; ++mt) {
      const bf16* asrow = &As[0][mt * 16 + l16][lh * 16];
      v16bf afrag = cat8(*(const v8bf*)asrow, *(const v8bf*)(asrow + 8));
      acc[mt] = wmma_bf16(afrag, bfrag, acc[mt]);
    }
    __syncthreads();
  }
#endif

  const int col = (int)n0 + w * 16 + l16;
  const float bcol = bias[col];
#pragma unroll
  for (int mt = 0; mt < 4; ++mt) {
#pragma unroll
    for (int r = 0; r < 8; ++r) {
      long row = m0 + mt * 16 + r + lh * 8;
      long o = row * (long)Nout + col;
      float v = acc[mt][r] + bcol;
      if (EPI == 0) {
        outb[o] = (bf16)v;
      } else if (EPI == 1) {
        v = 0.5f * v * (1.0f + erff(v * 0.70710678118654752f));
        outb[o] = (bf16)v;
      } else if (EPI == 2) {
        outf[o] = v;
      } else {
        outf[o] = v + res[o];
      }
    }
  }
}

// ---------------- fused attention: one block per (batch, head) ----------------
// LDS: K [224][64] bf16 | V^T [64][224] bf16 | per-wave P [16][224] bf16 (x8)

__global__ __launch_bounds__(256) void attention_k(const bf16* __restrict__ qkv,
                                                   bf16* __restrict__ o) {
  extern __shared__ char smem[];
  bf16* Ksm = (bf16*)smem;              // [224][64]
  bf16* Vt  = Ksm + 224 * 64;           // [64][224]
  bf16* Pw  = Vt + 64 * 224;            // 8 x [16][224]

  const int b = blockIdx.x / HEADS_, hh = blockIdx.x % HEADS_;
  const int tid = threadIdx.x, w = tid >> 5, lane = tid & 31;
  const int lh = lane >> 4, l16 = lane & 15;
  const long qbase = (long)b * N_ * (3 * D_) + hh * HD_;

  for (int idx = tid; idx < 224 * 64; idx += 256) {
    int n = idx >> 6, d = idx & 63;
    bf16 kv = (bf16)0.f, vv = (bf16)0.f;
    if (n < N_) {
      long p = qbase + (long)n * (3 * D_);
      kv = qkv[p + D_ + d];
      vv = qkv[p + 2 * D_ + d];
    }
    Ksm[n * 64 + d] = kv;
    Vt[d * 224 + n] = vv;
  }
  __syncthreads();

  bf16* P = Pw + w * (16 * 224);
  for (int qt = w; qt < 13; qt += 8) {
    int qrow = qt * 16 + l16;
    if (qrow > N_ - 1) qrow = N_ - 1;
    const bf16* qp = qkv + qbase + (long)qrow * (3 * D_);
    v16bf qa0 = cat8(*(const v8bf*)(qp + lh * 16),      *(const v8bf*)(qp + 8 + lh * 16));
    v16bf qa1 = cat8(*(const v8bf*)(qp + 32 + lh * 16), *(const v8bf*)(qp + 40 + lh * 16));

    v8f s[13] = {};
#pragma unroll
    for (int nt = 0; nt < 13; ++nt) {
      const bf16* kp0 = &Ksm[(nt * 16 + l16) * 64 + lh * 16];
      v16bf kb0 = cat8(*(const v8bf*)kp0, *(const v8bf*)(kp0 + 8));
      const bf16* kp1 = kp0 + 32;
      v16bf kb1 = cat8(*(const v8bf*)kp1, *(const v8bf*)(kp1 + 8));
      s[nt] = wmma_bf16(qa0, kb0, s[nt]);
      s[nt] = wmma_bf16(qa1, kb1, s[nt]);
    }

    // softmax along 208 cols (mask >=197); row lives in one VGPR across 16 lanes
#pragma unroll
    for (int r = 0; r < 8; ++r) {
      float rm = -1e30f;
#pragma unroll
      for (int nt = 0; nt < 13; ++nt) {
        int c = nt * 16 + l16;
        float val = (c < N_) ? s[nt][r] * 0.125f : -1e30f;
        s[nt][r] = val;
        rm = fmaxf(rm, val);
      }
#pragma unroll
      for (int off = 8; off; off >>= 1) rm = fmaxf(rm, __shfl_xor(rm, off, 16));
      float sum = 0.f;
#pragma unroll
      for (int nt = 0; nt < 13; ++nt) {
        float e = __expf(s[nt][r] - rm);
        s[nt][r] = e;
        sum += e;
      }
#pragma unroll
      for (int off = 8; off; off >>= 1) sum += __shfl_xor(sum, off, 16);
      float inv = 1.0f / sum;
      int prow = r + lh * 8;
#pragma unroll
      for (int nt = 0; nt < 13; ++nt)
        P[prow * 224 + nt * 16 + l16] = (bf16)(s[nt][r] * inv);
      P[prow * 224 + 208 + l16] = (bf16)0.f;  // zero pad cols 208..223
    }

    // O = P @ V  (K dim 224 zero-padded)
    v8f oacc[4] = {};
#pragma unroll
    for (int kt = 0; kt < 7; ++kt) {
      const bf16* pp = &P[l16 * 224 + kt * 32 + lh * 16];
      v16bf pa = cat8(*(const v8bf*)pp, *(const v8bf*)(pp + 8));
#pragma unroll
      for (int vt = 0; vt < 4; ++vt) {
        const bf16* vp = &Vt[(vt * 16 + l16) * 224 + kt * 32 + lh * 16];
        v16bf vb = cat8(*(const v8bf*)vp, *(const v8bf*)(vp + 8));
        oacc[vt] = wmma_bf16(pa, vb, oacc[vt]);
      }
    }
#pragma unroll
    for (int vt = 0; vt < 4; ++vt) {
#pragma unroll
      for (int r = 0; r < 8; ++r) {
        int m = qt * 16 + r + lh * 8;
        if (m < N_)
          o[(long)(b * N_ + m) * D_ + hh * HD_ + vt * 16 + l16] = (bf16)oacc[vt][r];
      }
    }
  }
}

// ---------------- host orchestration ----------------

extern "C" void kernel_launch(void* const* d_in, const int* in_sizes, int n_in,
                              void* d_out, int out_size, void* d_ws, size_t ws_size,
                              hipStream_t stream) {
  const float* x      = (const float*)d_in[0];
  const float* conv_w = (const float*)d_in[1];
  const float* conv_b = (const float*)d_in[2];
  const float* cls    = (const float*)d_in[3];
  const float* pos    = (const float*)d_in[4];
  const float* n1w    = (const float*)d_in[5];
  const float* n1b    = (const float*)d_in[6];
  const float* qkv_w  = (const float*)d_in[7];
  const float* qkv_b  = (const float*)d_in[8];
  const float* proj_w = (const float*)d_in[9];
  const float* proj_b = (const float*)d_in[10];
  const float* n2w    = (const float*)d_in[11];
  const float* n2b    = (const float*)d_in[12];
  const float* fc1_w  = (const float*)d_in[13];
  const float* fc1_b  = (const float*)d_in[14];
  const float* fc2_w  = (const float*)d_in[15];
  const float* fc2_b  = (const float*)d_in[16];
  const float* nfw    = (const float*)d_in[17];
  const float* nfb    = (const float*)d_in[18];

  char* ws = (char*)d_ws;
  size_t off = 0;
  auto alloc = [&](size_t bytes) -> char* {
    char* p = ws + off;
    off += (bytes + 255) & ~(size_t)255;
    return p;
  };

  bf16* convw_bf = (bf16*)alloc((size_t)D_ * D_ * 2);
  bf16* qkvw_bf  = (bf16*)alloc((size_t)DEPTH_ * 3 * D_ * D_ * 2);
  bf16* projw_bf = (bf16*)alloc((size_t)DEPTH_ * D_ * D_ * 2);
  bf16* fc1w_bf  = (bf16*)alloc((size_t)DEPTH_ * HID_ * D_ * 2);
  bf16* fc2w_bf  = (bf16*)alloc((size_t)DEPTH_ * D_ * HID_ * 2);
  float* h       = (float*)alloc((size_t)M_ * D_ * 4);
  bf16* hn       = (bf16*)alloc((size_t)M_ * D_ * 2);
  bf16* qkvb     = (bf16*)alloc((size_t)M_ * 3 * D_ * 2);
  bf16* ob       = (bf16*)alloc((size_t)M_ * D_ * 2);
  bf16* mlpb     = (bf16*)alloc((size_t)M_ * HID_ * 2);
  // im2col A and patch-GEMM tmp overlay the MLP buffer (dead before MLP runs)
  bf16* Abf   = (bf16*)mlpb;
  float* tmpf = (float*)((char*)mlpb + (((size_t)MP_ * D_ * 2 + 255) & ~(size_t)255));

  // 1) weights -> bf16
  cvt_bf16<<<2048, 256, 0, stream>>>(conv_w, convw_bf, D_ * D_);
  cvt_bf16<<<4096, 256, 0, stream>>>(qkv_w, qkvw_bf, DEPTH_ * 3 * D_ * D_);
  cvt_bf16<<<4096, 256, 0, stream>>>(proj_w, projw_bf, DEPTH_ * D_ * D_);
  cvt_bf16<<<4096, 256, 0, stream>>>(fc1_w, fc1w_bf, DEPTH_ * HID_ * D_);
  cvt_bf16<<<4096, 256, 0, stream>>>(fc2_w, fc2w_bf, DEPTH_ * D_ * HID_);

  // 2) patch embed: im2col + GEMM(K=768) + assemble(cls,pos)
  im2col_patch<<<4096, 256, 0, stream>>>(x, Abf);
  gemm_bf16_k<2><<<dim3(MP_ / 64, D_ / 128), 256, 0, stream>>>(
      Abf, convw_bf, conv_b, nullptr, tmpf, nullptr, D_, D_);
  assemble_tokens<<<4096, 256, 0, stream>>>(tmpf, cls, pos, h);

  // 3) transformer blocks
  for (int l = 0; l < DEPTH_; ++l) {
    layernorm_k<1><<<M_ / 8, 256, 0, stream>>>(h, n1w + l * D_, n1b + l * D_, hn, nullptr);
    gemm_bf16_k<0><<<dim3(M_ / 64, (3 * D_) / 128), 256, 0, stream>>>(
        hn, qkvw_bf + (size_t)l * 3 * D_ * D_, qkv_b + l * 3 * D_, nullptr,
        nullptr, qkvb, D_, 3 * D_);
    attention_k<<<B_ * HEADS_, 256, (224 * 64 + 64 * 224 + 8 * 16 * 224) * 2, stream>>>(qkvb, ob);
    gemm_bf16_k<3><<<dim3(M_ / 64, D_ / 128), 256, 0, stream>>>(
        ob, projw_bf + (size_t)l * D_ * D_, proj_b + l * D_, h, h, nullptr, D_, D_);
    layernorm_k<1><<<M_ / 8, 256, 0, stream>>>(h, n2w + l * D_, n2b + l * D_, hn, nullptr);
    gemm_bf16_k<1><<<dim3(M_ / 64, HID_ / 128), 256, 0, stream>>>(
        hn, fc1w_bf + (size_t)l * HID_ * D_, fc1_b + l * HID_, nullptr,
        nullptr, mlpb, D_, HID_);
    gemm_bf16_k<3><<<dim3(M_ / 64, D_ / 128), 256, 0, stream>>>(
        mlpb, fc2w_bf + (size_t)l * D_ * HID_, fc2_b + l * D_, h, h, nullptr, HID_, D_);
  }

  // 4) final LN -> f32 output
  layernorm_k<0><<<M_ / 8, 256, 0, stream>>>(h, nfw, nfb, nullptr, (float*)d_out);
}
